// Irregular_Convolution_LSTM_78323023610547
// MI455X (gfx1250) — matrix operand order repaired and checked by reading, not compiled, not run on hardware
//
#include <hip/hip_runtime.h>
#include <hip/hip_bf16.h>
#include <math.h>

// ---------------- problem constants ----------------
#define NB 8          // batch
#define NN 1000       // nodes
#define KK 16         // neighbors
#define HH 256        // lstm hidden
#define C1 32
#define C2 16
#define EPSV 1e-5f
#define BT_TOTAL 352  // branch T's 24/12/8 -> BT offsets {0,192,288}

typedef __bf16 bf16;
typedef __attribute__((ext_vector_type(16))) __bf16 v16bf;
typedef __attribute__((ext_vector_type(8)))  __bf16 v8bf;
typedef __attribute__((ext_vector_type(8)))  float  v8f;

__device__ inline v8f wmma_bf16(v16bf a, v16bf b, v8f c) {
  return __builtin_amdgcn_wmma_f32_16x16x32_bf16(false, a, false, b, (short)0, c, false, false);
}
__device__ inline v16bf cat16(v8bf lo, v8bf hi) {
  return __builtin_shufflevector(lo, hi, 0,1,2,3,4,5,6,7,8,9,10,11,12,13,14,15);
}

// ISA 7.12.2 wave32 layouts:
// A (16x32 bf16): row = lane&15; lane<16 K {0..7,16..23}; lane>=16 K {8..15,24..31}
__device__ inline int a_kidx(int half, int j) {
  return (j < 8) ? (half * 8 + j) : (16 + half * 8 + (j - 8));
}
// B (32x16 bf16): col = lane&15; k = a_kidx(half, j)  (pre-packed, so only used by packer)
__device__ inline float sigf(float x) { return 1.0f / (1.0f + __expf(-x)); }

__device__ inline void branch_of_bt(int bt, int& br, int& T, int& off, int& toff) {
  if (bt < 192)      { br = 0; T = 24; off = 0;   toff = 0;  }
  else if (bt < 288) { br = 1; T = 12; off = 192; toff = 24; }
  else               { br = 2; T = 8;  off = 288; toff = 36; }
}

// ---------------- TDM staging: global -> LDS (async tensor path) ---------
typedef unsigned int tdm_v4u __attribute__((ext_vector_type(4)));
typedef int          tdm_v8i __attribute__((ext_vector_type(8)));
typedef int          tdm_v4i __attribute__((ext_vector_type(4)));

// 1-D tile load: nelem elements of (1<<dsz_code) bytes, contiguous.
// D# per ISA 8.3/8.4: group0 {count=1, lds_addr, global_addr, type=2},
// group1 {data_size, tensor_dim0=tile_dim0=nelem, tensor_dim1=tile_dim1=1, stride0=nelem}.
__device__ inline void stage_to_lds(void* ldsdst, const void* gsrc,
                                    unsigned nelem, unsigned dsz_code,
                                    unsigned nbytes) {
#if __has_builtin(__builtin_amdgcn_tensor_load_to_lds) && __has_builtin(__builtin_amdgcn_s_wait_tensorcnt)
  if (threadIdx.x < 32) {  // one wave issues (TDM ignores EXEC)
    unsigned long long ga = (unsigned long long)gsrc;
    tdm_v4u g0;
    g0[0] = 1u;                                                  // count=1
    g0[1] = (unsigned)(unsigned long long)ldsdst;                // lds_addr
    g0[2] = (unsigned)ga;                                        // global_addr lo
    g0[3] = (unsigned)((ga >> 32) & 0x01FFFFFFu) | (2u << 30);   // ga hi | type=2
    tdm_v8i g1;
    g1[0] = (int)(dsz_code << 16);                 // data_size
    g1[1] = (int)((nelem & 0xFFFFu) << 16);        // tensor_dim0[15:0]
    g1[2] = (int)((nelem >> 16) | (1u << 16));     // tensor_dim0[31:16] | tensor_dim1=1
    g1[3] = (int)((nelem & 0xFFFFu) << 16);        // tile_dim0
    g1[4] = 1;                                     // tile_dim1 = 1
    g1[5] = (int)nelem;                            // tensor_dim0_stride lo
    g1[6] = 0; g1[7] = 0;
    tdm_v4i z4 = {0, 0, 0, 0};
#if defined(__clang_major__) && (__clang_major__ >= 23)
    tdm_v8i z8 = {0, 0, 0, 0, 0, 0, 0, 0};
    __builtin_amdgcn_tensor_load_to_lds(g0, g1, z4, z4, z8, 0);
#else
    __builtin_amdgcn_tensor_load_to_lds(g0, g1, z4, z4, 0);
#endif
    __builtin_amdgcn_s_wait_tensorcnt(0);
  }
#else
  const unsigned* s = (const unsigned*)gsrc;
  unsigned* d = (unsigned*)ldsdst;
  for (unsigned e = threadIdx.x; e < nbytes / 4; e += blockDim.x) d[e] = s[e];
#endif
  __syncthreads();
}

// ---------------- K0: zero init (output + BN stats) ----------------
__global__ void k0_init(float* out, float* stats) {
  int t = blockIdx.x * blockDim.x + threadIdx.x;
  if (t < NB * NN) out[t] = 0.0f;
  if (t < 44 * 16 * 2) stats[t] = 0.0f;
}

// ---------------- KP0: gather-index table -----------------------------
// gidx[f] = (f/16000)*1000 + nbr[f%16000]  (raw-reshape gather, shared by conv2/conv3)
__global__ void kp_gidx(const int* __restrict__ nbr, int* __restrict__ gidx) {
  int e = blockIdx.x * blockDim.x + threadIdx.x;
  if (e < NN * 512) {
    int cc = e / (NN * KK);
    int rr = e - cc * (NN * KK);
    gidx[e] = cc * NN + nbr[rr];
  }
}

// ---------------- KP1: pack f32 weights -> bf16 WMMA B-lane layout ------
// dst[e], e = ((ot*kbn + kb)*32 + lane)*16 + j ; k = kb*32 + a_kidx(lane>>4, j)
__global__ void kp_packw(const float* __restrict__ src, bf16* __restrict__ dst,
                         int Kdim, int kbn, int total) {
  for (int e = blockIdx.x * blockDim.x + threadIdx.x; e < total;
       e += gridDim.x * blockDim.x) {
    int j = e & 15, lane = (e >> 4) & 31;
    int kb = (e >> 9) % kbn, ot = e / (512 * kbn);
    int col = lane & 15, half = lane >> 4;
    int k = kb * 32 + a_kidx(half, j);
    float v = (k < Kdim) ? src[(ot * 16 + col) * Kdim + k] : 0.0f;
    dst[e] = (bf16)v;
  }
}

// ---------------- K1: conv1 + ReLU -> g1 (BT,32,N) bf16 ----------------
__global__ __launch_bounds__(32)
void k1_conv1(const float* __restrict__ xc, const float* __restrict__ xp,
              const float* __restrict__ xt, const float* __restrict__ W1,
              const float* __restrict__ b1, const int* __restrict__ nbr,
              bf16* __restrict__ g1) {
  int bt = blockIdx.y;
  int br, T, off, toff; branch_of_bt(bt, br, T, off, toff);
  const float* x = (br == 0) ? xc : (br == 1 ? xp : xt);
  int loc = bt - off, b = loc / T, t = loc % T;
  int lane = threadIdx.x, half = lane >> 4, col = lane & 15, row = lane & 15;
  int tile = blockIdx.x * 16;

  v16bf bw0, bw1;
#pragma unroll
  for (int j = 0; j < 16; ++j) {
    int k = a_kidx(half, j);
    float w0 = (k < KK) ? W1[br * 512 + col * KK + k] : 0.0f;
    float w1 = (k < KK) ? W1[br * 512 + (col + 16) * KK + k] : 0.0f;
    bw0[j] = (bf16)w0; bw1[j] = (bf16)w1;
  }
  v16bf a;
  int node = tile + row;
#pragma unroll
  for (int j = 0; j < 16; ++j) {
    int k = a_kidx(half, j);
    float v = 0.0f;
    if (k < KK && node < NN) v = x[(b * T + t) * NN + nbr[node * KK + k]];
    a[j] = (bf16)v;
  }
  v8f c0 = {}; v8f c1 = {};
  c0 = wmma_bf16(a, bw0, c0);
  c1 = wmma_bf16(a, bw1, c1);
#pragma unroll
  for (int r = 0; r < 8; ++r) {
    int nd = tile + r + (half << 3);
    if (nd < NN) {
      float v0 = c0[r] + b1[br * 32 + col];       v0 = v0 > 0.f ? v0 : 0.f;
      float v1 = c1[r] + b1[br * 32 + col + 16];  v1 = v1 > 0.f ? v1 : 0.f;
      g1[((long)bt * 32 + col) * NN + nd]      = (bf16)v0;
      g1[((long)bt * 32 + col + 16) * NN + nd] = (bf16)v1;
    }
  }
}

// ---------------- K2: conv2 (LDS-staged gather) -> g2 f32 + BN stats ----
// 8 waves/block share one TDM-staged 64KB g1 slab; B pre-packed bf16.
__global__ __launch_bounds__(256)
void k2_conv2(const bf16* __restrict__ g1, const bf16* __restrict__ W2p,
              const float* __restrict__ b2, const int* __restrict__ gidx,
              float* __restrict__ g2, float* __restrict__ stats) {
  extern __shared__ char smem[];
  bf16* g1s = (bf16*)smem;  // 32*1000 bf16 = 32,000 elem = 64,000 B
  int bt = blockIdx.y;
  int br, T, off, toff; branch_of_bt(bt, br, T, off, toff);
  int loc = bt - off, t = loc % T;
  stage_to_lds(g1s, g1 + (long)bt * 32 * NN, 32 * NN, /*2B*/1, 32 * NN * 2);

  int tid = threadIdx.x, lane = tid & 31, wave = tid >> 5;
  int half = lane >> 4, col = lane & 15, row = lane & 15;
  int tile = (blockIdx.x * 8 + wave) * 16;
  if (tile >= NN) return;  // wave-uniform
  int node = tile + row;

  v8f acc = {};
  for (int kb = 0; kb < 16; ++kb) {
    v16bf a;
#pragma unroll
    for (int j = 0; j < 16; ++j) {
      int k = kb * 32 + a_kidx(half, j);
      int f = (node < NN) ? node * 512 + k : 0;
      bf16 av = g1s[gidx[f]];
      a[j] = (node < NN) ? av : (bf16)0.0f;
    }
    const v8bf* bp = (const v8bf*)(W2p + (long)br * 8192 + kb * 512 + lane * 16);
    acc = wmma_bf16(a, cat16(bp[0], bp[1]), acc);
  }
  float ps = 0.f, pq = 0.f;
#pragma unroll
  for (int r = 0; r < 8; ++r) {
    int nd = tile + r + (half << 3);
    if (nd < NN) {
      float v = acc[r] + b2[br * 16 + col];
      g2[((long)bt * 16 + col) * NN + nd] = v;
      ps += v; pq += v * v;
    }
  }
  ps += __shfl_xor(ps, 16);
  pq += __shfl_xor(pq, 16);
  if (lane < 16) {
    int sidx = ((toff + t) * 16 + col) * 2;
    atomicAdd(&stats[sidx + 0], ps);
    atomicAdd(&stats[sidx + 1], pq);
  }
}

// ---------------- K3: BN + ReLU + conv3 + ReLU -> sbuf bf16 (R,1024) ----
__global__ __launch_bounds__(256)
void k3_bn_conv3(const float* __restrict__ g2, const float* __restrict__ stats,
                 const float* __restrict__ gamma, const float* __restrict__ beta,
                 const float* __restrict__ W3, const float* __restrict__ b3,
                 const int* __restrict__ gidx, bf16* __restrict__ sbuf) {
  extern __shared__ char smem[];
  float* g2s = (float*)smem;  // 16*1000 f32 = 64,000 B
  int bt = blockIdx.x;
  int br, T, off, toff; branch_of_bt(bt, br, T, off, toff);
  int loc = bt - off, b = loc / T, t = loc % T;
  stage_to_lds(g2s, g2 + (long)bt * 16 * NN, 16 * NN, /*4B*/2, 16 * NN * 4);

  float scl[C2], sft[C2];
#pragma unroll
  for (int c = 0; c < C2; ++c) {
    float su = stats[((toff + t) * 16 + c) * 2 + 0];
    float sq = stats[((toff + t) * 16 + c) * 2 + 1];
    float mu = su / (float)(NB * NN);
    float var = sq / (float)(NB * NN) - mu * mu;
    float inv = rsqrtf(var + EPSV);
    scl[c] = gamma[br * 16 + c] * inv;
    sft[c] = beta[br * 16 + c] - mu * scl[c];
  }
  int sOffE[3] = {0, 196608, 294912};
  long rowbase = sOffE[br] + (long)(t * NB + b) * 1024;
  for (int it = 0; it < 4; ++it) {
    int node = threadIdx.x + 256 * it;  // covers 0..1023 (1000..1023 = zero pad)
    float outv = 0.0f;
    if (node < NN) {
      float acc = 0.f;
      for (int j = 0; j < C2 * KK; ++j) {
        int f = node * (C2 * KK) + j;
        int cc = f / (NN * KK);
        float h = g2s[gidx[f]] * scl[cc] + sft[cc];
        h = h > 0.f ? h : 0.f;
        acc += h * W3[br * 256 + j];
      }
      acc += b3[br];
      outv = acc > 0.f ? acc : 0.f;
    }
    sbuf[rowbase + node] = (bf16)outv;
  }
}

// ---------------- K4: LSTM layer-0 input projection GEMM ---------------
// P[t*B+b, o] = s_row @ Wih0^T + bih0 + bhh0 ; all-b128 bf16 loads
__global__ __launch_bounds__(32)
void k4_inproj(const bf16* __restrict__ sbuf, const bf16* __restrict__ Wih0p,
               const float* __restrict__ bih0, const float* __restrict__ bhh0,
               float* __restrict__ P) {
  int br = blockIdx.z;
  int Ts[3] = {24, 12, 8};
  int R = Ts[br] * NB;
  int tile = blockIdx.x * 16;
  if (tile >= R) return;
  int sOffE[3] = {0, 196608, 294912};
  int pOff[3]  = {0, 196608, 294912};
  int lane = threadIdx.x, half = lane >> 4, col = lane & 15, row = lane & 15;
  int ot = blockIdx.y;  // 0..63
  const bf16* A = sbuf + sOffE[br] + (long)(tile + row) * 1024;
  const bf16* W = Wih0p + (long)br * 1048576 + (long)ot * (32 * 512);

  v8f acc = {};
  for (int kb = 0; kb < 32; ++kb) {
    __builtin_prefetch(W + (kb + 2) * 512 + lane * 16, 0, 3);
    const v8bf* ap = (const v8bf*)(A + kb * 32 + half * 8);
    const v8bf* bp = (const v8bf*)(W + kb * 512 + lane * 16);
    v16bf av = cat16(ap[0], *(const v8bf*)(A + kb * 32 + half * 8 + 16));
    acc = wmma_bf16(av, cat16(bp[0], bp[1]), acc);
  }
  int o = ot * 16 + col;
  float bias = bih0[br * 1024 + o] + bhh0[br * 1024 + o];
#pragma unroll
  for (int r = 0; r < 8; ++r)
    P[pOff[br] + (long)(tile + r + (half << 3)) * 1024 + o] = acc[r] + bias;
}

// ---------------- K5: recurrence + FC + fuse (one WGP per branch) -------
__device__ inline v8f mm256_pk(const bf16* __restrict__ A /*LDS 16x256*/,
                               const bf16* __restrict__ Wp /*packed, otile base*/,
                               int lane, v8f acc) {
  int half = lane >> 4, row = lane & 15;
  for (int kb = 0; kb < 8; ++kb) {
    __builtin_prefetch(Wp + (kb + 2) * 512 + lane * 16, 0, 3);
    const v8bf* ap = (const v8bf*)(A + row * HH + kb * 32 + half * 8);
    const v8bf* bp = (const v8bf*)(Wp + kb * 512 + lane * 16);
    v16bf av = cat16(ap[0], *(const v8bf*)(A + row * HH + kb * 32 + half * 8 + 16));
    acc = wmma_bf16(av, cat16(bp[0], bp[1]), acc);
  }
  return acc;
}

__global__ __launch_bounds__(256)
void k5_lstm(const float* __restrict__ P, const bf16* __restrict__ Whh0p,
             const bf16* __restrict__ Wih1p, const bf16* __restrict__ Whh1p,
             const float* __restrict__ bih1, const float* __restrict__ bhh1,
             const float* __restrict__ fc1W, const float* __restrict__ fc1b,
             const float* __restrict__ fc2W, const float* __restrict__ fc2b,
             const float* __restrict__ fuse, float* __restrict__ out) {
  __shared__ __attribute__((aligned(16))) bf16 h0bf[16 * HH];
  __shared__ __attribute__((aligned(16))) bf16 h1bf[16 * HH];
  __shared__ float c0s[NB * HH];
  __shared__ float c1s[NB * HH];
  __shared__ float gbuf[NB * 1024];

  int br = blockIdx.x;
  int Ts[3] = {24, 12, 8};
  int pOff[3] = {0, 196608, 294912};
  int T = Ts[br];
  const float* Pb  = P     + pOff[br];
  const bf16* W0h  = Whh0p + (long)br * 262144;
  const bf16* W1i  = Wih1p + (long)br * 262144;
  const bf16* W1h  = Whh1p + (long)br * 262144;

  int tid = threadIdx.x, lane = tid & 31, wave = tid >> 5;
  int half = lane >> 4, col = lane & 15;

  for (int e = tid; e < 16 * HH; e += 256) { h0bf[e] = (bf16)0.f; h1bf[e] = (bf16)0.f; }
  for (int e = tid; e < NB * HH; e += 256) { c0s[e] = 0.f; c1s[e] = 0.f; }

  for (int t = 0; t < T; ++t) {
    __syncthreads();
    // layer 0 gates: P(t) + h0 @ Whh0^T
    for (int wt = 0; wt < 8; ++wt) {
      int ot = wave * 8 + wt;
      v8f acc;
#pragma unroll
      for (int r = 0; r < 8; ++r) {
        int row = r + (half << 3);
        int rc = row < NB ? row : 0;
        float pv = Pb[(long)(t * NB + rc) * 1024 + (ot * 16 + col)];
        acc[r] = (row < NB) ? pv : 0.0f;
      }
      acc = mm256_pk(h0bf, W0h + ot * 4096, lane, acc);
#pragma unroll
      for (int r = 0; r < 8; ++r) {
        int row = r + (half << 3);
        if (row < NB) gbuf[row * 1024 + ot * 16 + col] = acc[r];
      }
    }
    __syncthreads();
    for (int e = tid; e < NB * HH; e += 256) {
      int b = e >> 8, hx = e & 255;
      float gi = gbuf[b * 1024 + hx],       gf = gbuf[b * 1024 + 256 + hx];
      float gg = gbuf[b * 1024 + 512 + hx], go = gbuf[b * 1024 + 768 + hx];
      float c = sigf(gf) * c0s[e] + sigf(gi) * tanhf(gg);
      c0s[e] = c;
      h0bf[b * HH + hx] = (bf16)(sigf(go) * tanhf(c));
    }
    __syncthreads();
    // layer 1 gates: ys @ Wih1^T + h1 @ Whh1^T + biases
    for (int wt = 0; wt < 8; ++wt) {
      int ot = wave * 8 + wt;
      float bias = bih1[br * 1024 + ot * 16 + col] + bhh1[br * 1024 + ot * 16 + col];
      v8f acc;
#pragma unroll
      for (int r = 0; r < 8; ++r) acc[r] = bias;
      acc = mm256_pk(h0bf, W1i + ot * 4096, lane, acc);
      acc = mm256_pk(h1bf, W1h + ot * 4096, lane, acc);
#pragma unroll
      for (int r = 0; r < 8; ++r) {
        int row = r + (half << 3);
        if (row < NB) gbuf[row * 1024 + ot * 16 + col] = acc[r];
      }
    }
    __syncthreads();
    for (int e = tid; e < NB * HH; e += 256) {
      int b = e >> 8, hx = e & 255;
      float gi = gbuf[b * 1024 + hx],       gf = gbuf[b * 1024 + 256 + hx];
      float gg = gbuf[b * 1024 + 512 + hx], go = gbuf[b * 1024 + 768 + hx];
      float c = sigf(gf) * c1s[e] + sigf(gi) * tanhf(gg);
      c1s[e] = c;
      h1bf[b * HH + hx] = (bf16)(sigf(go) * tanhf(c));
    }
  }
  __syncthreads();
  for (int e = tid; e < NB * 64; e += 256) {   // fc1 -> gbuf[0..511]
    int b = e >> 6, u = e & 63;
    float acc = fc1b[br * 64 + u];
    for (int k = 0; k < HH; ++k)
      acc += (float)h1bf[b * HH + k] * fc1W[((long)br * 64 + u) * HH + k];
    gbuf[e] = acc;
  }
  __syncthreads();
  for (int e = tid; e < NB * NN; e += 256) {   // fc2 + tanh + fuse
    int b = e / NN, n = e - b * NN;
    float acc = fc2b[br * NN + n];
    for (int u = 0; u < 64; ++u)
      acc += gbuf[b * 64 + u] * fc2W[((long)br * NN + n) * 64 + u];
    atomicAdd(&out[b * NN + n], tanhf(acc) * fuse[br * NN + n]);
  }
}

// ---------------- launcher ----------------
extern "C" void kernel_launch(void* const* d_in, const int* in_sizes, int n_in,
                              void* d_out, int out_size, void* d_ws, size_t ws_size,
                              hipStream_t stream) {
  const float* xc    = (const float*)d_in[0];
  const float* xp    = (const float*)d_in[1];
  const float* xt    = (const float*)d_in[2];
  const float* W1    = (const float*)d_in[3];
  const float* b1    = (const float*)d_in[4];
  const float* W2    = (const float*)d_in[5];
  const float* b2    = (const float*)d_in[6];
  const float* gamma = (const float*)d_in[7];
  const float* beta  = (const float*)d_in[8];
  const float* W3    = (const float*)d_in[9];
  const float* b3    = (const float*)d_in[10];
  const float* Wih0  = (const float*)d_in[11];
  const float* Whh0  = (const float*)d_in[12];
  const float* bih0  = (const float*)d_in[13];
  const float* bhh0  = (const float*)d_in[14];
  const float* Wih1  = (const float*)d_in[15];
  const float* Whh1  = (const float*)d_in[16];
  const float* bih1  = (const float*)d_in[17];
  const float* bhh1  = (const float*)d_in[18];
  const float* fc1W  = (const float*)d_in[19];
  const float* fc1b  = (const float*)d_in[20];
  const float* fc2W  = (const float*)d_in[21];
  const float* fc2b  = (const float*)d_in[22];
  const float* fuse  = (const float*)d_in[23];
  const int*   nbr   = (const int*)d_in[24];
  float* out = (float*)d_out;

  // workspace carve (bytes, all 16B-aligned):
  char* ws = (char*)d_ws;
  bf16*  g1    = (bf16*)(ws);                    // 352*32*1000 bf16 = 22,528,000
  float* g2    = (float*)(ws + 22528000);        // 352*16*1000 f32  = 22,528,000
  float* stats = (float*)(ws + 45056000);        // 44*16*2 f32
  bf16*  sbuf  = (bf16*)(ws + 45059584);         // 352*1024 bf16 = 720,896
  float* Pbuf  = (float*)(ws + 45780480);        // 352*1024 f32  = 1,441,792
  int*   gidx  = (int*)(ws + 47222272);          // 512,000 int   = 2,048,000
  bf16*  W2p   = (bf16*)(ws + 49270272);         // 3*8192 bf16
  bf16*  Wih0p = (bf16*)(ws + 49319424);         // 3*1,048,576 bf16
  bf16*  Whh0p = (bf16*)(ws + 55610880);         // 3*262,144 bf16
  bf16*  Wih1p = (bf16*)(ws + 57183744);         // 3*262,144 bf16
  bf16*  Whh1p = (bf16*)(ws + 58756608);         // 3*262,144 bf16

  k0_init<<<dim3(37), dim3(256), 0, stream>>>(out, stats);
  kp_gidx<<<dim3(2000), dim3(256), 0, stream>>>(nbr, gidx);
  for (int br = 0; br < 3; ++br) {
    kp_packw<<<dim3(32), dim3(256), 0, stream>>>(W2 + br * 8192, W2p + br * 8192,
                                                 512, 16, 8192);
    kp_packw<<<dim3(1024), dim3(256), 0, stream>>>(Wih0 + (long)br * 1024 * NN,
                                                   Wih0p + (long)br * 1048576,
                                                   1000, 32, 1048576);
    kp_packw<<<dim3(512), dim3(256), 0, stream>>>(Whh0 + (long)br * 1024 * HH,
                                                  Whh0p + (long)br * 262144,
                                                  256, 8, 262144);
    kp_packw<<<dim3(512), dim3(256), 0, stream>>>(Wih1 + (long)br * 1024 * HH,
                                                  Wih1p + (long)br * 262144,
                                                  256, 8, 262144);
    kp_packw<<<dim3(512), dim3(256), 0, stream>>>(Whh1 + (long)br * 1024 * HH,
                                                  Whh1p + (long)br * 262144,
                                                  256, 8, 262144);
  }
  k1_conv1<<<dim3(63, BT_TOTAL), dim3(32), 0, stream>>>(xc, xp, xt, W1, b1, nbr, g1);
  k2_conv2<<<dim3(8, BT_TOTAL), dim3(256), 128000, stream>>>(g1, W2p, b2, gidx, g2, stats);
  k3_bn_conv3<<<dim3(BT_TOTAL), dim3(256), 64000, stream>>>(g2, stats, gamma, beta,
                                                            W3, b3, gidx, sbuf);
  k4_inproj<<<dim3(12, 64, 3), dim3(32), 0, stream>>>(sbuf, Wih0p, bih0, bhh0, Pbuf);
  k5_lstm<<<dim3(3), dim3(256), 0, stream>>>(Pbuf, Whh0p, Wih1p, Whh1p, bih1, bhh1,
                                             fc1W, fc1b, fc2W, fc2b, fuse, out);
}
